// MultiHeadLatentAttention_14139032338627
// MI455X (gfx1250) — matrix operand order repaired and checked by reading, not compiled, not run on hardware
//
#include <hip/hip_runtime.h>
#include <math.h>

// ---------------------------------------------------------------------------
// MLA forward for MI455X (gfx1250, wave32, WMMA + async-to-LDS).
//   B=2 T=2048 HIDDEN=2048 HEADS=16 HEAD_DIM=128 ROT=64 LATENT=512
// GEMMs + attention run through v_wmma_f32_16x16x32_bf16 (fp32 accum);
// tile movement uses global_load_async_to_lds_b128 with double-buffered LDS.
// Steady-state loops are peeled so the async-wait is branch-free.
// ---------------------------------------------------------------------------

typedef __bf16 bf16_t;
typedef __attribute__((ext_vector_type(16))) __bf16 v16bf;
typedef __attribute__((ext_vector_type(8)))  __bf16 v8bf;
typedef __attribute__((ext_vector_type(8)))  float  v8f;

static constexpr int T_SEQ   = 2048;
static constexpr int BATCH   = 2;
static constexpr int HEADS   = 16;
static constexpr int HDIM    = 128;
static constexpr int HIDDEN  = 2048;
static constexpr int LATENT  = 512;
static constexpr int ROTD    = 1024;   // heads*64

__device__ __forceinline__ v16bf cat8(v8bf lo, v8bf hi) {
  return __builtin_shufflevector(lo, hi, 0,1,2,3,4,5,6,7,8,9,10,11,12,13,14,15);
}

// A fragment 16x32 bf16 from row-major [m][k] buffer, leading dim ld (elems).
// lane m=l&15, half=l>>4; elem i -> k = half*16 + i (matches B frags).
__device__ __forceinline__ v16bf load_a_frag(const bf16_t* base, int ld) {
  const int lane = threadIdx.x & 31;
  const int m = lane & 15, half = lane >> 4;
  return *(const v16bf*)(base + (size_t)m * ld + half * 16);
}

// B fragment 32x16 from B^T layout [n][k], leading dim ld:
// lane n=l&15, half=l>>4; elem i -> k = half*16 + i  (contiguous 32B).
__device__ __forceinline__ v16bf load_bt_frag(const bf16_t* base, int ld) {
  const int lane = threadIdx.x & 31;
  const int n = lane & 15, half = lane >> 4;
  return *(const v16bf*)(base + (size_t)n * ld + half * 16);
}

// ---- CDNA5 async global->LDS copy (ASYNCcnt tracked, bypasses VGPRs) ------
__device__ __forceinline__ void async_b128(unsigned lds_off, const bf16_t* g) {
  asm volatile("global_load_async_to_lds_b128 %0, %1, off"
               :: "v"(lds_off), "v"(g) : "memory");
}
// copy 64B (32 bf16) per lane
__device__ __forceinline__ void async_copy64(bf16_t* lds, const bf16_t* g) {
  unsigned off = (unsigned)(unsigned long long)(uintptr_t)lds; // flat addr low 32 = LDS offset
  async_b128(off,      g);
  async_b128(off + 16, g + 8);
  async_b128(off + 32, g + 16);
  async_b128(off + 48, g + 24);
}
__device__ __forceinline__ void async_wait0() {
  asm volatile("s_wait_asynccnt 0x0" ::: "memory");
}
__device__ __forceinline__ void async_wait8() {
  asm volatile("s_wait_asynccnt 0x8" ::: "memory");
}

// ---------------------------------------------------------------------------
// fp32 -> bf16 elementwise
__global__ void cvt_bf16_kernel(const float* __restrict__ src,
                                bf16_t* __restrict__ dst, int n) {
  int i = blockIdx.x * blockDim.x + threadIdx.x;
  int stride = gridDim.x * blockDim.x;
  for (; i < n; i += stride) dst[i] = (bf16_t)src[i];
}

// fp32 [K][N] -> bf16 [N][K]   (weight transpose so GEMM B-frags are contiguous)
__global__ void cvt_tr_bf16_kernel(const float* __restrict__ src,
                                   bf16_t* __restrict__ dst, int K, int N) {
  int i = blockIdx.x * blockDim.x + threadIdx.x;
  int total = K * N;
  int stride = gridDim.x * blockDim.x;
  for (; i < total; i += stride) {
    int n = i / K, k = i - n * K;
    dst[i] = (bf16_t)src[(size_t)k * N + n];
  }
}

// ---------------------------------------------------------------------------
// C[M,N] = A[M,K] @ B[K,N];  A bf16 row-major, Bt bf16 [N][K].
// 128x128 block tile, 8 waves (4x2), wave = 32x64 = 2x4 WMMA tiles.
// K-tile 64, double-buffered LDS fed by async-to-LDS copies; last K-step peeled.
__launch_bounds__(256)
__global__ void gemm_bf16_wmma(const bf16_t* __restrict__ A,
                               const bf16_t* __restrict__ Bt,
                               bf16_t* __restrict__ Cb,
                               float*  __restrict__ Cf,
                               int M, int N, int K) {
  __shared__ alignas(64) bf16_t Alds[2][128 * 64];
  __shared__ alignas(64) bf16_t Blds[2][128 * 64];

  const int tid   = threadIdx.x;
  const int wave  = tid >> 5;
  const int waveM = wave & 3;   // 32-row group
  const int waveN = wave >> 2;  // 64-col group
  const int mBase = blockIdx.y * 128;
  const int nBase = blockIdx.x * 128;

  const v8f z8 = {0.f,0.f,0.f,0.f,0.f,0.f,0.f,0.f};
  v8f acc[2][4];
#pragma unroll
  for (int a = 0; a < 2; ++a)
#pragma unroll
    for (int b = 0; b < 4; ++b) acc[a][b] = z8;

  // tile-copy assignment: thread -> (row, 32-elem chunk)
  const int r = tid >> 1;
  const int c = (tid & 1) * 32;
  const bf16_t* gA = A  + (size_t)(mBase + r) * K + c;
  const bf16_t* gB = Bt + (size_t)(nBase + r) * K + c;

  // one 64-K-step of WMMAs on buffer `cur` (B frags preloaded for ILP)
  auto compute = [&](int cur) {
#pragma unroll
    for (int ks = 0; ks < 2; ++ks) {
      v16bf af0 = load_a_frag(&Alds[cur][(waveM * 32 +  0) * 64 + ks * 32], 64);
      v16bf af1 = load_a_frag(&Alds[cur][(waveM * 32 + 16) * 64 + ks * 32], 64);
      v16bf bfr[4];
#pragma unroll
      for (int nt = 0; nt < 4; ++nt)
        bfr[nt] = load_bt_frag(&Blds[cur][(waveN * 64 + nt * 16) * 64 + ks * 32], 64);
#pragma unroll
      for (int nt = 0; nt < 4; ++nt) {
        acc[0][nt] = __builtin_amdgcn_wmma_f32_16x16x32_bf16(
            false, af0, false, bfr[nt], (short)0, acc[0][nt], false, false);
        acc[1][nt] = __builtin_amdgcn_wmma_f32_16x16x32_bf16(
            false, af1, false, bfr[nt], (short)0, acc[1][nt], false, false);
      }
    }
  };

  const int kSteps = K >> 6;            // >= 8 for all our shapes
  // prologue: tile 0 -> buffer 0 (8 async b128 per thread per tile)
  async_copy64(&Alds[0][r * 64 + c], gA);
  async_copy64(&Blds[0][r * 64 + c], gB);

  for (int kt = 0; kt + 1 < kSteps; ++kt) {   // steady state: branch-free wait
    const int cur = kt & 1;
    const size_t kOff = (size_t)(kt + 1) << 6;
    async_copy64(&Alds[cur ^ 1][r * 64 + c], gA + kOff);
    async_copy64(&Blds[cur ^ 1][r * 64 + c], gB + kOff);
    async_wait8();                      // in-order: tile kt's 8 ops complete
    __syncthreads();                    // tile kt visible to all waves
    compute(cur);
    __syncthreads();                    // buffer reuse safe
  }
  async_wait0();                        // peeled last tile
  __syncthreads();
  compute((kSteps - 1) & 1);

  const int lane = tid & 31;
  const int ln = lane & 15, half = lane >> 4;
  if (Cf) {
#pragma unroll
    for (int mt = 0; mt < 2; ++mt)
#pragma unroll
      for (int nt = 0; nt < 4; ++nt)
#pragma unroll
        for (int v = 0; v < 8; ++v) {
          const int row = mBase + waveM * 32 + mt * 16 + v + 8 * half;
          const int col = nBase + waveN * 64 + nt * 16 + ln;
          Cf[(size_t)row * N + col] = acc[mt][nt][v];
        }
  }
  if (Cb) {
#pragma unroll
    for (int mt = 0; mt < 2; ++mt)
#pragma unroll
      for (int nt = 0; nt < 4; ++nt)
#pragma unroll
        for (int v = 0; v < 8; ++v) {
          const int row = mBase + waveM * 32 + mt * 16 + v + 8 * half;
          const int col = nBase + waveN * 64 + nt * 16 + ln;
          Cb[(size_t)row * N + col] = (bf16_t)acc[mt][nt][v];
        }
  }
}

// ---------------------------------------------------------------------------
// RoPE + head packing:
//   Qp[b,h,t,d] = d<64 ? rope(Qrot) : Q[...,d];   Kp likewise
//   Vt[b,h,d,t] = V[b,t,h,d]                      (B^T layout for P@V WMMA)
__global__ void rope_pack_kernel(const bf16_t* __restrict__ Q,
                                 const bf16_t* __restrict__ Kf,
                                 const bf16_t* __restrict__ V,
                                 const bf16_t* __restrict__ Qr,
                                 const bf16_t* __restrict__ Kr,
                                 bf16_t* __restrict__ Qp,
                                 bf16_t* __restrict__ Kp,
                                 bf16_t* __restrict__ Vt) {
  const int bt = blockIdx.x;               // 0..B*T-1
  const int b = bt / T_SEQ, t = bt - b * T_SEQ;
  const size_t rowIn  = (size_t)bt * HIDDEN;
  const size_t rowRot = (size_t)bt * ROTD;

  for (int idx = threadIdx.x; idx < HIDDEN; idx += blockDim.x) {
    const int h = idx >> 7, d = idx & 127;
    float qv, kv;
    if (d < 64) {
      const int j = d & 31;
      // inv_freq = 10000^(-j/32) = exp(-j * ln(10000)/32)
      const float inv = __expf(-(float)j * 0.28782313662425572f);
      const float ang = (float)t * inv;
      const float cs = __cosf(ang), sn = __sinf(ang);
      const size_t rb = rowRot + (size_t)h * 64;
      const float q1 = (float)Qr[rb + d];
      const float k1 = (float)Kr[rb + d];
      if (d < 32) {
        qv = q1 * cs - (float)Qr[rb + d + 32] * sn;
        kv = k1 * cs - (float)Kr[rb + d + 32] * sn;
      } else {
        qv = q1 * cs + (float)Qr[rb + d - 32] * sn;
        kv = k1 * cs + (float)Kr[rb + d - 32] * sn;
      }
    } else {
      qv = (float)Q[rowIn + idx];
      kv = (float)Kf[rowIn + idx];
    }
    const size_t o = (((size_t)b * HEADS + h) * T_SEQ + t) * HDIM + d;
    Qp[o] = (bf16_t)qv;
    Kp[o] = (bf16_t)kv;
    Vt[(((size_t)b * HEADS + h) * HDIM + d) * T_SEQ + t] = V[rowIn + idx];
  }
}

// ---------------------------------------------------------------------------
// Causal flash attention. Grid: (T/128, B*H). 8 waves; each wave owns 16 query
// rows. Key blocks of 64 with double-buffered async K/V tiles; online softmax;
// WMMA for QK^T and P@V; last key block peeled. Writes Y bf16 in [B*T, HIDDEN].
__launch_bounds__(256)
__global__ void flash_attn_kernel(const bf16_t* __restrict__ Qp,
                                  const bf16_t* __restrict__ Kp,
                                  const bf16_t* __restrict__ Vt,
                                  bf16_t* __restrict__ Y) {
  __shared__ alignas(64) bf16_t Klds[2][64 * 128];    // [key][d]
  __shared__ alignas(64) bf16_t Vlds[2][128 * 64];    // [d][key]  (B^T layout)
  __shared__ alignas(64) bf16_t Plds[8 * 16 * 64];    // per-wave P staging

  const int tid  = threadIdx.x;
  const int wave = tid >> 5;
  const int lane = tid & 31;
  const int ln = lane & 15, half = lane >> 4;
  const int bh = blockIdx.y;
  const int b = bh >> 4, h = bh & 15;
  const int qBlock = blockIdx.x * 128;
  const int qBase  = qBlock + wave * 16;

  const bf16_t* Qhead = Qp + (size_t)bh * T_SEQ * HDIM;
  const bf16_t* Khead = Kp + (size_t)bh * T_SEQ * HDIM;
  const bf16_t* Vhead = Vt + (size_t)bh * HDIM * T_SEQ;

  // async K/V tile issue (8 b128 per thread per tile)
  const int kr = tid >> 2, kc = (tid & 3) * 32;
  const int vd = tid >> 1, vc = (tid & 1) * 32;
  auto issueKV = [&](int kbase, int buf) {
    async_copy64(&Klds[buf][kr * 128 + kc],
                 Khead + (size_t)(kbase + kr) * HDIM + kc);
    async_copy64(&Vlds[buf][vd * 64 + vc],
                 Vhead + (size_t)vd * T_SEQ + kbase + vc);
  };

  // Q fragments for this wave's 16 rows (d = 4 K-steps of 32)
  v16bf qfrag[4];
#pragma unroll
  for (int ks = 0; ks < 4; ++ks) {
    const bf16_t* p = Qhead + (size_t)(qBase + ln) * HDIM + ks * 32 + half * 16;
    qfrag[ks] = *(const v16bf*)p;
  }

  const v8f z8 = {0.f,0.f,0.f,0.f,0.f,0.f,0.f,0.f};
  v8f o[8];
  float mrow[8], lrow[8];
#pragma unroll
  for (int dt = 0; dt < 8; ++dt) o[dt] = z8;
#pragma unroll
  for (int v = 0; v < 8; ++v) { mrow[v] = -1e30f; lrow[v] = 0.f; }

  const float scale = 0.08838834764831845f;   // 1/sqrt(128)
  bf16_t* Pw = &Plds[wave * 16 * 64];

  // one key-block step (tile data already waited on by caller)
  auto attn_step = [&](int it) {
    const int kb = it << 6;
    const int cur = it & 1;
    __syncthreads();                          // tile kb visible

    // S = Q @ K^T  (4 column tiles of 16; B frags preloaded per k-substep)
    v8f s[4];
#pragma unroll
    for (int nt = 0; nt < 4; ++nt) s[nt] = z8;
#pragma unroll
    for (int ks = 0; ks < 4; ++ks) {
      v16bf bfr[4];
#pragma unroll
      for (int nt = 0; nt < 4; ++nt)
        bfr[nt] = *(const v16bf*)&Klds[cur][(nt * 16 + ln) * 128 + ks * 32 + half * 16];
#pragma unroll
      for (int nt = 0; nt < 4; ++nt)
        s[nt] = __builtin_amdgcn_wmma_f32_16x16x32_bf16(
            false, qfrag[ks], false, bfr[nt], (short)0, s[nt], false, false);
    }

    // scale + causal mask + per-tile row max
    float tmax[8];
#pragma unroll
    for (int v = 0; v < 8; ++v) tmax[v] = -1e30f;
#pragma unroll
    for (int nt = 0; nt < 4; ++nt) {
      const int col = kb + nt * 16 + ln;
#pragma unroll
      for (int v = 0; v < 8; ++v) {
        const int row = qBase + v + 8 * half;
        float val = s[nt][v] * scale;
        val = (col <= row) ? val : -1e30f;
        s[nt][v] = val;
        tmax[v] = fmaxf(tmax[v], val);
      }
    }
    // row max across the 16-lane group holding the row's columns
#pragma unroll
    for (int v = 0; v < 8; ++v)
#pragma unroll
      for (int off = 8; off > 0; off >>= 1)
        tmax[v] = fmaxf(tmax[v], __shfl_xor(tmax[v], off, 32));

    float corr[8], rsum[8];
#pragma unroll
    for (int v = 0; v < 8; ++v) {
      const float mnew = fmaxf(mrow[v], tmax[v]);
      corr[v] = __expf(mrow[v] - mnew);
      mrow[v] = mnew;
      rsum[v] = 0.f;
    }
#pragma unroll
    for (int nt = 0; nt < 4; ++nt)
#pragma unroll
      for (int v = 0; v < 8; ++v) {
        const float p = __expf(s[nt][v] - mrow[v]);
        s[nt][v] = p;
        rsum[v] += p;
      }
#pragma unroll
    for (int v = 0; v < 8; ++v) {
#pragma unroll
      for (int off = 8; off > 0; off >>= 1)
        rsum[v] += __shfl_xor(rsum[v], off, 32);
      lrow[v] = lrow[v] * corr[v] + rsum[v];
#pragma unroll
      for (int dt = 0; dt < 8; ++dt) o[dt][v] *= corr[v];
    }

    // P -> LDS (A-fragment layout needs [row][key] row-major)
#pragma unroll
    for (int nt = 0; nt < 4; ++nt)
#pragma unroll
      for (int v = 0; v < 8; ++v)
        Pw[(v + 8 * half) * 64 + nt * 16 + ln] = (bf16_t)s[nt][v];
    __syncthreads();

    // O += P @ V  (V frags preloaded in groups of 4 d-tiles)
#pragma unroll
    for (int ks = 0; ks < 2; ++ks) {
      const bf16_t* pp = Pw + ln * 64 + ks * 32 + half * 16;
      v16bf pfrag = *(const v16bf*)pp;
#pragma unroll
      for (int g = 0; g < 2; ++g) {
        v16bf vfr[4];
#pragma unroll
        for (int j = 0; j < 4; ++j)
          vfr[j] = *(const v16bf*)&Vlds[cur][((g * 4 + j) * 16 + ln) * 64 + ks * 32 + half * 16];
#pragma unroll
        for (int j = 0; j < 4; ++j)
          o[g * 4 + j] = __builtin_amdgcn_wmma_f32_16x16x32_bf16(
              false, pfrag, false, vfr[j], (short)0, o[g * 4 + j], false, false);
      }
    }
    __syncthreads();                          // buffer `cur` free for reuse
  };

  const int nIter = (qBlock + 128) >> 6;      // key blocks of 64 (>= 2)
  issueKV(0, 0);                              // prologue
  for (int it = 0; it + 1 < nIter; ++it) {    // steady state: branch-free wait
    issueKV((it + 1) << 6, (it + 1) & 1);
    async_wait8();
    attn_step(it);
  }
  async_wait0();                              // peeled last key block
  attn_step(nIter - 1);

  // normalize + write Y[b, t, h*128 + d]
#pragma unroll
  for (int v = 0; v < 8; ++v) {
    const float inv = 1.0f / lrow[v];
    const int t = qBase + v + 8 * half;
    const size_t rowOff = ((size_t)b * T_SEQ + t) * HIDDEN + h * HDIM;
#pragma unroll
    for (int dt = 0; dt < 8; ++dt)
      Y[rowOff + dt * 16 + ln] = (bf16_t)(o[dt][v] * inv);
  }
}

// ---------------------------------------------------------------------------
extern "C" void kernel_launch(void* const* d_in, const int* in_sizes, int n_in,
                              void* d_out, int out_size, void* d_ws, size_t ws_size,
                              hipStream_t stream) {
  const float* x     = (const float*)d_in[0];
  const float* Wkv_d = (const float*)d_in[1];
  const float* Wq_d  = (const float*)d_in[2];
  const float* Wk_u  = (const float*)d_in[3];
  const float* Wq_u  = (const float*)d_in[4];
  const float* Wv_u  = (const float*)d_in[5];
  const float* Wrq   = (const float*)d_in[6];
  const float* Wrk   = (const float*)d_in[7];
  const float* Wo    = (const float*)d_in[8];
  float* out = (float*)d_out;

  const int M = BATCH * T_SEQ;           // 4096
  size_t off = 0;
  auto alloc = [&](size_t elems) -> bf16_t* {
    bf16_t* p = (bf16_t*)d_ws + off;
    off += (elems + 127) & ~(size_t)127;  // keep 256B alignment
    return p;
  };
  bf16_t* xb   = alloc((size_t)M * HIDDEN);
  bf16_t* wkvd = alloc((size_t)LATENT * HIDDEN);   // [512][2048]
  bf16_t* wqd  = alloc((size_t)LATENT * HIDDEN);
  bf16_t* wku  = alloc((size_t)HIDDEN * LATENT);   // [2048][512]
  bf16_t* wqu  = alloc((size_t)HIDDEN * LATENT);
  bf16_t* wvu  = alloc((size_t)HIDDEN * LATENT);
  bf16_t* wrq  = alloc((size_t)ROTD * LATENT);     // [1024][512]
  bf16_t* wrk  = alloc((size_t)ROTD * HIDDEN);     // [1024][2048]
  bf16_t* wo   = alloc((size_t)HIDDEN * HIDDEN);
  bf16_t* kvl  = alloc((size_t)M * LATENT);
  bf16_t* ql   = alloc((size_t)M * LATENT);
  bf16_t* Kf   = alloc((size_t)M * HIDDEN);
  bf16_t* Qf   = alloc((size_t)M * HIDDEN);
  bf16_t* Vf   = alloc((size_t)M * HIDDEN);
  bf16_t* Qr   = alloc((size_t)M * ROTD);
  bf16_t* Kr   = alloc((size_t)M * ROTD);
  bf16_t* Qp   = alloc((size_t)M * HIDDEN);
  bf16_t* Kp   = alloc((size_t)M * HIDDEN);
  bf16_t* Vt   = alloc((size_t)M * HIDDEN);
  bf16_t* Yb   = alloc((size_t)M * HIDDEN);
  (void)ws_size;

  // 1) conversions
  cvt_bf16_kernel<<<4096, 256, 0, stream>>>(x, xb, M * HIDDEN);
  cvt_tr_bf16_kernel<<<2048, 256, 0, stream>>>(Wkv_d, wkvd, HIDDEN, LATENT);
  cvt_tr_bf16_kernel<<<2048, 256, 0, stream>>>(Wq_d,  wqd,  HIDDEN, LATENT);
  cvt_tr_bf16_kernel<<<2048, 256, 0, stream>>>(Wk_u,  wku,  LATENT, HIDDEN);
  cvt_tr_bf16_kernel<<<2048, 256, 0, stream>>>(Wq_u,  wqu,  LATENT, HIDDEN);
  cvt_tr_bf16_kernel<<<2048, 256, 0, stream>>>(Wv_u,  wvu,  LATENT, HIDDEN);
  cvt_tr_bf16_kernel<<<2048, 256, 0, stream>>>(Wrq,   wrq,  LATENT, ROTD);
  cvt_tr_bf16_kernel<<<2048, 256, 0, stream>>>(Wrk,   wrk,  HIDDEN, ROTD);
  cvt_tr_bf16_kernel<<<2048, 256, 0, stream>>>(Wo,    wo,   HIDDEN, HIDDEN);

  // 2) projection GEMMs (grid = N/128 x M/128)
  gemm_bf16_wmma<<<dim3(LATENT/128, M/128), 256, 0, stream>>>(xb,  wkvd, kvl, nullptr, M, LATENT, HIDDEN);
  gemm_bf16_wmma<<<dim3(LATENT/128, M/128), 256, 0, stream>>>(xb,  wqd,  ql,  nullptr, M, LATENT, HIDDEN);
  gemm_bf16_wmma<<<dim3(HIDDEN/128, M/128), 256, 0, stream>>>(kvl, wku,  Kf,  nullptr, M, HIDDEN, LATENT);
  gemm_bf16_wmma<<<dim3(HIDDEN/128, M/128), 256, 0, stream>>>(ql,  wqu,  Qf,  nullptr, M, HIDDEN, LATENT);
  gemm_bf16_wmma<<<dim3(HIDDEN/128, M/128), 256, 0, stream>>>(kvl, wvu,  Vf,  nullptr, M, HIDDEN, LATENT);
  gemm_bf16_wmma<<<dim3(ROTD/128,   M/128), 256, 0, stream>>>(ql,  wrq,  Qr,  nullptr, M, ROTD,   LATENT);
  gemm_bf16_wmma<<<dim3(ROTD/128,   M/128), 256, 0, stream>>>(xb,  wrk,  Kr,  nullptr, M, ROTD,   HIDDEN);

  // 3) RoPE + pack to per-head layouts
  rope_pack_kernel<<<M, 256, 0, stream>>>(Qf, Kf, Vf, Qr, Kr, Qp, Kp, Vt);

  // 4) causal flash attention -> Y (bf16, [B*T, HIDDEN])
  flash_attn_kernel<<<dim3(T_SEQ/128, BATCH*HEADS), 256, 0, stream>>>(Qp, Kp, Vt, Yb);

  // 5) output projection (fp32 out)
  gemm_bf16_wmma<<<dim3(HIDDEN/128, M/128), 256, 0, stream>>>(Yb, wo, nullptr, out, M, HIDDEN, HIDDEN);
}
